// Attention_42872363549181
// MI455X (gfx1250) — matrix operand order repaired and checked by reading, not compiled, not run on hardware
//
#include <hip/hip_runtime.h>

// ---------------------------------------------------------------------------
// Types / WMMA helpers (CDNA5 gfx1250, wave32)
// ---------------------------------------------------------------------------
typedef __bf16 bf16_t;
typedef __attribute__((ext_vector_type(16))) __bf16 v16bf;
typedef __attribute__((ext_vector_type(8)))  __bf16 v8bf;
typedef __attribute__((ext_vector_type(8)))  float  v8f;
typedef __attribute__((ext_vector_type(4))) unsigned int u32x4;
typedef __attribute__((ext_vector_type(4))) int          i32x4;
typedef __attribute__((ext_vector_type(8))) int          i32x8;

static constexpr int BATCH = 2;
static constexpr int SEQ   = 2048;
static constexpr int DIM   = 1024;
static constexpr int HEADS = 16;
static constexpr int HD    = 64;          // head dim
static constexpr int MROWS = BATCH * SEQ; // 4096
static constexpr float ATT_SCALE = 0.125f; // 64^-0.5

// LDS tile geometry for the TDM-staged GEMMs: 128 rows x 32 bf16, with TDM
// padding of 4 DWORDs every 16 DWORDs -> row stride 40 elements (80B), which
// spreads fragment reads across LDS banks.
static constexpr int KSTEP   = 32;
static constexpr int LDS_LDA = 40;                 // padded row stride (elems)
static constexpr int TILE_E  = 128 * LDS_LDA;      // elems per staged tile

__device__ __forceinline__ v8f wmma_bf16(v16bf a, v16bf b, v8f c) {
  // D = A(16x32 bf16) * B(32x16 bf16) + C(16x16 f32)
  return __builtin_amdgcn_wmma_f32_16x16x32_bf16(
      /*neg_a=*/false, a, /*neg_b=*/false, b,
      /*c_mod=*/(short)0, c, /*reuse_a=*/false, /*reuse_b=*/false);
}

// A-frag: 16x32 (MxK) row-major source. lane<16: M=lane, K={0..7,16..23};
// lane>=16: M=lane-16, K={8..15,24..31}. Two 16B contiguous loads per lane.
__device__ __forceinline__ v16bf load_a_frag(const bf16_t* base, int ld, int lane) {
  const int row = lane & 15;
  const int kh  = lane >> 4;
  const bf16_t* p = base + (size_t)row * ld + kh * 8;
  v8bf lo = *(const v8bf*)(p);
  v8bf hi = *(const v8bf*)(p + 16);
  v16bf a;
#pragma unroll
  for (int i = 0; i < 8; ++i) { a[i] = lo[i]; a[i + 8] = hi[i]; }
  return a;
}

// B-frag: 32x16 (KxN) where source memory is [N][K] row-major (i.e. B^T rows
// contiguous). lane<16: N=lane, K={0..7,16..23}; lane>=16: N=lane-16, K={8..15,24..31}.
__device__ __forceinline__ v16bf load_b_frag(const bf16_t* base, int ld, int lane) {
  const int col = lane & 15;
  const int kh  = lane >> 4;
  const bf16_t* p = base + (size_t)col * ld + kh * 8;
  v8bf lo = *(const v8bf*)(p);
  v8bf hi = *(const v8bf*)(p + 16);
  v16bf b;
#pragma unroll
  for (int i = 0; i < 8; ++i) { b[i] = lo[i]; b[i + 8] = hi[i]; }
  return b;
}

__device__ __forceinline__ v8f zero_v8f() {
  v8f z = {0.f, 0.f, 0.f, 0.f, 0.f, 0.f, 0.f, 0.f};
  return z;
}

// ---------------------------------------------------------------------------
// Tensor Data Mover: 2D tile load global -> LDS.
//   tile: tile_rows x tile_cols bf16 elements, source rows strided by
//   row_stride elems; LDS destination gets 4 DWORDs of padding every 16 DWORDs
//   (row = 32 elems = 16 DWORDs -> padded LDS row stride = 40 elems).
// Issued by one wave; completion tracked with TENSORcnt.
// ---------------------------------------------------------------------------
__device__ __forceinline__ void tdm_load_2d(bf16_t* lds_dst, const bf16_t* gsrc,
                                            unsigned tile_rows, unsigned tile_cols,
                                            unsigned row_stride, unsigned tensor_rows) {
  const unsigned lds_addr = (unsigned)(size_t)lds_dst;          // LDS byte offset
  const unsigned long long ga = (unsigned long long)(size_t)gsrc;

  u32x4 g0;
  g0[0] = 1u;                                  // count=1, no gather
  g0[1] = lds_addr;                            // [63:32] lds_addr
  g0[2] = (unsigned)(ga & 0xffffffffu);        // [95:64] global_addr lo
  g0[3] = (unsigned)((ga >> 32) & 0x1ffffffu)  // [120:96] global_addr hi
        | (2u << 30);                          // [127:126] type=2 (image)

  const unsigned td0 = row_stride;             // tensor dim0 (elems per row)
  const unsigned td1 = tensor_rows;            // tensor dim1 (rows)
  const unsigned long long st0 = row_stride;   // dim0 stride (elems)

  i32x8 g1;
  g1[0] = (int)((1u << 16)                     // data_size=1 -> 2 bytes
        | (1u << 20)                           // pad_enable
        | (3u << 22)                           // pad_interval: 16 DWORDs
        | (3u << 25));                         // pad_amount:   4 DWORDs
  g1[1] = (int)((td0 & 0xffffu) << 16);                         // tensor_dim0 lo
  g1[2] = (int)(((td0 >> 16) & 0xffffu) | ((td1 & 0xffffu) << 16));
  g1[3] = (int)(((td1 >> 16) & 0xffffu) | (tile_cols << 16));   // tile_dim0
  g1[4] = (int)(tile_rows & 0xffffu);                           // tile_dim1
  g1[5] = (int)(unsigned)(st0 & 0xffffffffu);                   // dim0_stride lo
  g1[6] = (int)(unsigned)((st0 >> 32) & 0xffffu);               // dim0_stride hi
  g1[7] = 0;

  i32x4 z4 = {0, 0, 0, 0};
#if defined(__clang_major__) && (__clang_major__ >= 23)
  i32x8 z8 = {0, 0, 0, 0, 0, 0, 0, 0};
  __builtin_amdgcn_tensor_load_to_lds(g0, g1, z4, z4, z8, 0);
#else
  __builtin_amdgcn_tensor_load_to_lds(g0, g1, z4, z4, 0);
#endif
}

// ---------------------------------------------------------------------------
// Kernel 0: fp32 -> bf16 conversion (memory-bound, trivial vs. 23.3 TB/s)
// ---------------------------------------------------------------------------
__global__ void cvt_f32_bf16(const float* __restrict__ src, bf16_t* __restrict__ dst, int n) {
  int i = blockIdx.x * blockDim.x + threadIdx.x;
  if (i < n) dst[i] = (bf16_t)src[i];
}

// ---------------------------------------------------------------------------
// Block-tile GEMM core: 128x128 tile per 8-wave workgroup, K in steps of 32.
// TDM double-buffers A(128x32) and B(128x32) bf16 tiles into LDS; each wave
// computes a 32x64 sub-tile with 8 WMMAs per K-step from LDS fragments.
// Returns per-wave accumulators; caller does the epilogue.
// ---------------------------------------------------------------------------
struct GemmAcc { v8f acc[2][4]; int m0, n0, lane; };

__device__ __forceinline__ void gemm_block_tdm(
    GemmAcc& g, const bf16_t* __restrict__ A, const bf16_t* __restrict__ Bm,
    int K, int a_rows, int b_rows, int bm, int bn,
    bf16_t* ldsA, bf16_t* ldsB) {
  const int lane = threadIdx.x & 31;
  const int wv   = threadIdx.x >> 5;                    // 0..7
  const int blkM = bm * 128, blkN = bn * 128;
  const int wrow = (wv >> 1) * 32;                      // wave row in block
  const int wcol = (wv & 1) * 64;                       // wave col in block
  g.lane = lane;
  g.m0 = blkM + wrow;
  g.n0 = blkN + wcol;
#pragma unroll
  for (int mi = 0; mi < 2; ++mi)
#pragma unroll
    for (int ni = 0; ni < 4; ++ni) g.acc[mi][ni] = zero_v8f();

  // prologue: stage K-step 0 into buffer 0
  if (wv == 0) {
    tdm_load_2d(ldsA, A + (size_t)blkM * K, 128, KSTEP, (unsigned)K, (unsigned)a_rows);
    tdm_load_2d(ldsB, Bm + (size_t)blkN * K, 128, KSTEP, (unsigned)K, (unsigned)b_rows);
    __builtin_amdgcn_s_wait_tensorcnt(0);
  }
  __syncthreads();

  for (int k0 = 0; k0 < K; k0 += KSTEP) {
    const int buf = (k0 >> 5) & 1;
    const bf16_t* tA = ldsA + buf * TILE_E;
    const bf16_t* tB = ldsB + buf * TILE_E;
    // prefetch next K-step into the other buffer (its readers finished at the
    // barrier that ended the previous iteration)
    if (wv == 0 && (k0 + KSTEP) < K) {
      bf16_t* nA = ldsA + (buf ^ 1) * TILE_E;
      bf16_t* nB = ldsB + (buf ^ 1) * TILE_E;
      tdm_load_2d(nA, A + (size_t)blkM * K + (k0 + KSTEP), 128, KSTEP, (unsigned)K, (unsigned)a_rows);
      tdm_load_2d(nB, Bm + (size_t)blkN * K + (k0 + KSTEP), 128, KSTEP, (unsigned)K, (unsigned)b_rows);
    }

    v16bf a0 = load_a_frag(tA + (size_t)wrow * LDS_LDA, LDS_LDA, lane);
    v16bf a1 = load_a_frag(tA + (size_t)(wrow + 16) * LDS_LDA, LDS_LDA, lane);
#pragma unroll
    for (int ni = 0; ni < 4; ++ni) {
      v16bf bf = load_b_frag(tB + (size_t)(wcol + ni * 16) * LDS_LDA, LDS_LDA, lane);
      g.acc[0][ni] = wmma_bf16(a0, bf, g.acc[0][ni]);
      g.acc[1][ni] = wmma_bf16(a1, bf, g.acc[1][ni]);
    }

    if (wv == 0 && (k0 + KSTEP) < K) __builtin_amdgcn_s_wait_tensorcnt(0);
    __syncthreads();
  }
}

// ---------------------------------------------------------------------------
// Kernel 1: QKV GEMM  qkv[m,d] = sum_k x[m,k] * w_qkv[d,k]
// Epilogue scatters into Q[B,H,N,hd] (pre-scaled), K[B,H,N,hd], V^T[B,H,hd,N].
// ---------------------------------------------------------------------------
__global__ void qkv_gemm(const bf16_t* __restrict__ xb, const bf16_t* __restrict__ wb,
                         bf16_t* __restrict__ Qb, bf16_t* __restrict__ Kb,
                         bf16_t* __restrict__ Vt) {
  __shared__ bf16_t ldsA[2 * TILE_E];
  __shared__ bf16_t ldsB[2 * TILE_E];
  const int tilesN = (3 * DIM) / 128;                  // 24
  const int bm = blockIdx.x / tilesN;
  const int bn = blockIdx.x % tilesN;

  GemmAcc g;
  gemm_block_tdm(g, xb, wb, DIM, MROWS, 3 * DIM, bm, bn, ldsA, ldsB);

  const int half = g.lane >> 4, nl = g.lane & 15;
#pragma unroll
  for (int mi = 0; mi < 2; ++mi)
#pragma unroll
    for (int ni = 0; ni < 4; ++ni)
#pragma unroll
      for (int v = 0; v < 8; ++v) {
        const int m = g.m0 + mi * 16 + v + 8 * half;
        const int d = g.n0 + ni * 16 + nl;
        const float val = g.acc[mi][ni][v];
        const int s   = d >> 10;        // 0=Q 1=K 2=V
        const int rem = d & 1023;
        const int h   = rem >> 6;
        const int e   = rem & 63;
        const int b   = m >> 11;        // / SEQ
        const int n   = m & 2047;
        const int bh  = b * HEADS + h;
        if (s == 0)      Qb[((size_t)bh * SEQ + n) * HD + e] = (bf16_t)(val * ATT_SCALE);
        else if (s == 1) Kb[((size_t)bh * SEQ + n) * HD + e] = (bf16_t)val;
        else             Vt[((size_t)bh * HD + e) * SEQ + n] = (bf16_t)val;
      }
}

// ---------------------------------------------------------------------------
// Kernel 2: flash attention. One wave per (b,h, 16-query tile); 32-key blocks.
// S = Q K^T via 4 WMMAs, online softmax (shfl_xor width-16 row reductions),
// P transposed C-layout -> A-layout via a 1KB per-wave LDS round trip,
// O += P V via 4 WMMAs against V^T.
// ---------------------------------------------------------------------------
__global__ void flash_attn(const bf16_t* __restrict__ Qb, const bf16_t* __restrict__ Kb,
                           const bf16_t* __restrict__ Vt, bf16_t* __restrict__ Ob) {
  __shared__ bf16_t plds[8 * 16 * 32];                // 8 waves * 16x32 P tile
  const int lane = threadIdx.x & 31;
  const int winb = threadIdx.x >> 5;
  const int w    = blockIdx.x * 8 + winb;
  const int numQT = SEQ / 16;                          // 128
  if (w >= BATCH * HEADS * numQT) return;
  const int bh = w / numQT;
  const int qt = w % numQT;
  bf16_t* pl = plds + winb * (16 * 32);

  const bf16_t* Qp = Qb + ((size_t)bh * SEQ + qt * 16) * HD;
  const bf16_t* Kp = Kb + (size_t)bh * SEQ * HD;
  const bf16_t* Vp = Vt + (size_t)bh * HD * SEQ;

  const v16bf qa0 = load_a_frag(Qp, HD, lane);        // hd 0..31
  const v16bf qa1 = load_a_frag(Qp + 32, HD, lane);   // hd 32..63

  v8f o[4];
#pragma unroll
  for (int t = 0; t < 4; ++t) o[t] = zero_v8f();
  float mrow[8], lrow[8];
#pragma unroll
  for (int v = 0; v < 8; ++v) { mrow[v] = -3.0e38f; lrow[v] = 0.f; }

  const int half = lane >> 4, nl = lane & 15;

  for (int kb = 0; kb < SEQ; kb += 32) {
    // ---- S = Q K^T for 32 keys (two 16x16 score tiles) ----
    v8f s0 = zero_v8f(), s1 = zero_v8f();
    {
      v16bf bk = load_b_frag(Kp + (size_t)kb * HD, HD, lane);
      s0 = wmma_bf16(qa0, bk, s0);
      bk = load_b_frag(Kp + (size_t)kb * HD + 32, HD, lane);
      s0 = wmma_bf16(qa1, bk, s0);
      bk = load_b_frag(Kp + (size_t)(kb + 16) * HD, HD, lane);
      s1 = wmma_bf16(qa0, bk, s1);
      bk = load_b_frag(Kp + (size_t)(kb + 16) * HD + 32, HD, lane);
      s1 = wmma_bf16(qa1, bk, s1);
    }
    // ---- online softmax over this 32-key block ----
    float alpha[8];
#pragma unroll
    for (int v = 0; v < 8; ++v) {
      float t = fmaxf(s0[v], s1[v]);
#pragma unroll
      for (int off = 8; off >= 1; off >>= 1)
        t = fmaxf(t, __shfl_xor(t, off, 16));
      const float mn = fmaxf(mrow[v], t);
      alpha[v] = __expf(mrow[v] - mn);
      mrow[v] = mn;
    }
#pragma unroll
    for (int v = 0; v < 8; ++v) {
      const float p0 = __expf(s0[v] - mrow[v]);
      const float p1 = __expf(s1[v] - mrow[v]);
      const int row = v + 8 * half;
      pl[row * 32 + nl]      = (bf16_t)p0;            // ds_store_b16
      pl[row * 32 + 16 + nl] = (bf16_t)p1;
      float t = p0 + p1;
#pragma unroll
      for (int off = 8; off >= 1; off >>= 1)
        t += __shfl_xor(t, off, 16);
      lrow[v] = lrow[v] * alpha[v] + t;
    }
#pragma unroll
    for (int t = 0; t < 4; ++t)
#pragma unroll
      for (int v = 0; v < 8; ++v) o[t][v] *= alpha[v];

    // ---- read P back as an A-frag (16x32) from LDS (intra-wave, DScnt) ----
    v16bf pa;
    {
      const bf16_t* pp = pl + nl * 32 + half * 8;
      v8bf lo = *(const v8bf*)(pp);
      v8bf hi = *(const v8bf*)(pp + 16);
#pragma unroll
      for (int i = 0; i < 8; ++i) { pa[i] = lo[i]; pa[i + 8] = hi[i]; }
    }
    // ---- O += P V : B-frags contiguous thanks to V^T layout ----
#pragma unroll
    for (int t = 0; t < 4; ++t) {
      v16bf bv = load_b_frag(Vp + (size_t)(t * 16) * SEQ + kb, SEQ, lane);
      o[t] = wmma_bf16(pa, bv, o[t]);
    }
  }

  // ---- normalize and store to O[B*N, C] (bf16 staging for proj GEMM) ----
  const int b = bh / HEADS, h = bh % HEADS;
#pragma unroll
  for (int t = 0; t < 4; ++t)
#pragma unroll
    for (int v = 0; v < 8; ++v) {
      const int row = qt * 16 + v + 8 * half;
      const size_t m = (size_t)b * SEQ + row;
      const int col = h * HD + t * 16 + nl;
      Ob[m * DIM + col] = (bf16_t)(o[t][v] / lrow[v]);
    }
}

// ---------------------------------------------------------------------------
// Kernel 3: output projection  out[m,d] = sum_k O[m,k]*w_proj[d,k] + b[d]
// ---------------------------------------------------------------------------
__global__ void proj_gemm(const bf16_t* __restrict__ Ob, const bf16_t* __restrict__ wb,
                          const float* __restrict__ bias, float* __restrict__ out) {
  __shared__ bf16_t ldsA[2 * TILE_E];
  __shared__ bf16_t ldsB[2 * TILE_E];
  const int tilesN = DIM / 128;                        // 8
  const int bm = blockIdx.x / tilesN;
  const int bn = blockIdx.x % tilesN;

  GemmAcc g;
  gemm_block_tdm(g, Ob, wb, DIM, MROWS, DIM, bm, bn, ldsA, ldsB);

  const int half = g.lane >> 4, nl = g.lane & 15;
#pragma unroll
  for (int mi = 0; mi < 2; ++mi)
#pragma unroll
    for (int ni = 0; ni < 4; ++ni)
#pragma unroll
      for (int v = 0; v < 8; ++v) {
        const int m = g.m0 + mi * 16 + v + 8 * half;
        const int d = g.n0 + ni * 16 + nl;
        out[(size_t)m * DIM + d] = g.acc[mi][ni][v] + bias[d];
      }
}

// ---------------------------------------------------------------------------
// Launcher
// ---------------------------------------------------------------------------
extern "C" void kernel_launch(void* const* d_in, const int* in_sizes, int n_in,
                              void* d_out, int out_size, void* d_ws, size_t ws_size,
                              hipStream_t stream) {
  const float* x      = (const float*)d_in[0];   // [2,2048,1024]
  const float* w_qkv  = (const float*)d_in[1];   // [3072,1024]
  const float* w_proj = (const float*)d_in[2];   // [1024,1024]
  const float* b_proj = (const float*)d_in[3];   // [1024]
  float* out = (float*)d_out;

  const size_t nX    = (size_t)MROWS * DIM;       // 4M
  const size_t nWqkv = (size_t)3 * DIM * DIM;     // 3M
  const size_t nWprj = (size_t)DIM * DIM;         // 1M
  const size_t nHead = (size_t)BATCH * HEADS * SEQ * HD; // 4M

  char* ws = (char*)d_ws;
  size_t off = 0;
  bf16_t* xb     = (bf16_t*)(ws + off); off += nX    * sizeof(bf16_t);
  bf16_t* wqkvb  = (bf16_t*)(ws + off); off += nWqkv * sizeof(bf16_t);
  bf16_t* wprojb = (bf16_t*)(ws + off); off += nWprj * sizeof(bf16_t);
  bf16_t* Qb     = (bf16_t*)(ws + off); off += nHead * sizeof(bf16_t);
  bf16_t* Kb     = (bf16_t*)(ws + off); off += nHead * sizeof(bf16_t);
  bf16_t* Vt     = (bf16_t*)(ws + off); off += nHead * sizeof(bf16_t);
  bf16_t* Ob     = (bf16_t*)(ws + off); off += nX    * sizeof(bf16_t);
  (void)ws_size; (void)in_sizes; (void)n_in; (void)out_size;

  // fp32 -> bf16 staging
  cvt_f32_bf16<<<(int)((nX    + 255) / 256), 256, 0, stream>>>(x,      xb,     (int)nX);
  cvt_f32_bf16<<<(int)((nWqkv + 255) / 256), 256, 0, stream>>>(w_qkv,  wqkvb,  (int)nWqkv);
  cvt_f32_bf16<<<(int)((nWprj + 255) / 256), 256, 0, stream>>>(w_proj, wprojb, (int)nWprj);

  // QKV projection: (4096/128) x (3072/128) = 32 x 24 block tiles
  qkv_gemm<<<32 * 24, 256, 0, stream>>>(xb, wqkvb, Qb, Kb, Vt);

  // Flash attention: 4096 waves, 8 waves/block
  flash_attn<<<4096 / 8, 256, 0, stream>>>(Qb, Kb, Vt, Ob);

  // Output projection: (4096/128) x (1024/128) = 32 x 8 block tiles
  proj_gemm<<<32 * 8, 256, 0, stream>>>(Ob, wprojb, b_proj, out);
}
